// SummaryNetwork_24979529794232
// MI455X (gfx1250) — compile-verified, matrix operand order checked by reading, not run on hardware
//
#include <hip/hip_runtime.h>
#include <hip/hip_bf16.h>
#include <stdint.h>

// ---------------- problem dims ----------------
#define B_   64
#define T_   512
#define E_   64
#define H_   128
#define D_   256     // 2*H
#define NH_  8
#define KD_  32
#define G4H  512     // 4*H (gate width)
#define DU_  256
#define OUT_ 128
#define NEGF (-1.0e9f)

typedef __attribute__((ext_vector_type(16))) __bf16 v16bf;
typedef __attribute__((ext_vector_type(8)))  __bf16 v8bf;
typedef __attribute__((ext_vector_type(8)))  float  v8f;

// Vectorized WMMA fragment loader for k_stride==1 sources.
// Per ISA 7.12.2 (16-bit 16x32 A / 32x16 B): lane li=lane&15 selects the
// major index (M row for A, N col for B); per-lane K values are the two
// contiguous 8-element runs [8*half, 8*half+8) and [16+8*half, 24+8*half).
// Requires: base 16B-aligned, lane_stride a multiple of 8 elements.
__device__ __forceinline__ v16bf load_frag_k1(const __bf16* __restrict__ base,
                                              int lane_stride) {
  const int lane = threadIdx.x & 31;
  const __bf16* p = base + (size_t)(lane & 15) * lane_stride + 8 * (lane >> 4);
  v8bf lo = *(const v8bf*)p;         // K run 0
  v8bf hi = *(const v8bf*)(p + 16);  // K run 1
  return __builtin_shufflevector(lo, hi, 0, 1, 2, 3, 4, 5, 6, 7,
                                 8, 9, 10, 11, 12, 13, 14, 15);
}

// ---------------- weight convert + transpose ----------------
// src: f32 [K][N]  ->  dst: bf16 [N][K]
__global__ void cvt_transpose_kernel(const float* __restrict__ src,
                                     __bf16* __restrict__ dst, int K, int N) {
  int i = blockIdx.x * 256 + threadIdx.x;
  if (i < K * N) {
    const int k = i / N, n = i % N;
    dst[(size_t)n * K + k] = (__bf16)src[i];
  }
}

// embedding gather + mask; block = 64 threads (one (b,t) per block)
__global__ void embed_kernel(const int* __restrict__ seq,
                             const float* __restrict__ emb,
                             __bf16* __restrict__ xbf,
                             unsigned char* __restrict__ mask) {
  const int bt = blockIdx.x;
  const int e  = threadIdx.x;
  const int s  = seq[bt];
  xbf[(size_t)bt * E_ + e] = (__bf16)emb[(size_t)s * E_ + e];
  if (e == 0) mask[bt] = (s != 0) ? 1 : 0;
}

// repack V: [B,T,NH,KD] -> [B,NH,KD,T] so attn B-fragments get k_stride 1
__global__ void vtrans_kernel(const __bf16* __restrict__ vbuf,
                              __bf16* __restrict__ vT) {
  const size_t i = (size_t)blockIdx.x * 256 + threadIdx.x;  // over B*T*D
  const int c  = (int)(i & (D_ - 1));
  const size_t bt = i >> 8;                // / D_
  const int t  = (int)(bt & (T_ - 1));
  const int b  = (int)(bt >> 9);           // / T_
  const int h  = c >> 5, kd = c & 31;
  vT[((size_t)(b * NH_ + h) * KD_ + kd) * T_ + t] = vbuf[i];
}

// ---------------- generic bf16 WMMA GEMM ----------------
// C[M,N] = A[M,K](bf16,row-major) @ B (given as BT[N][K], bf16) (+bias)(relu?)
// block = 256 (8 waves); wave w does tiles (mt=blockIdx.y, nt0=2*(bx*8+w), nt0+1)
__global__ void gemm_bf16_kernel(const __bf16* __restrict__ A,
                                 const __bf16* __restrict__ BT,
                                 const float* __restrict__ bias,
                                 __bf16* __restrict__ outBf,
                                 float* __restrict__ outF,
                                 int M, int N, int K, int relu) {
  const int w   = threadIdx.x >> 5;
  const int nt0 = (blockIdx.x * 8 + w) * 2;
  const int mt  = blockIdx.y;
  if (nt0 * 16 >= N) return;  // wave-uniform
  v8f acc0 = {0.f, 0.f, 0.f, 0.f, 0.f, 0.f, 0.f, 0.f};
  v8f acc1 = {0.f, 0.f, 0.f, 0.f, 0.f, 0.f, 0.f, 0.f};
  const __bf16* Abase = A  + (size_t)mt * 16 * K;
  const __bf16* B0    = BT + (size_t)(nt0 * 16) * K;
  const __bf16* B1    = B0 + (size_t)16 * K;
  for (int kk = 0; kk < K; kk += 32) {
    v16bf a  = load_frag_k1(Abase + kk, K);
    v16bf b0 = load_frag_k1(B0 + kk, K);
    acc0 = __builtin_amdgcn_wmma_f32_16x16x32_bf16(false, a, false, b0,
                                                   (short)0, acc0, false, false);
    v16bf b1 = load_frag_k1(B1 + kk, K);
    acc1 = __builtin_amdgcn_wmma_f32_16x16x32_bf16(false, a, false, b1,
                                                   (short)0, acc1, false, false);
  }
  const int lane = threadIdx.x & 31;
  const int half = lane >> 4;
#pragma unroll
  for (int r = 0; r < 8; ++r) {
    const int m  = mt * 16 + r + 8 * half;
    const int n0 = nt0 * 16 + (lane & 15);
    float v0 = acc0[r], v1 = acc1[r];
    if (bias) { v0 += bias[n0]; v1 += bias[n0 + 16]; }
    if (relu) { v0 = fmaxf(v0, 0.f); v1 = fmaxf(v1, 0.f); }
    if (outBf) {
      outBf[(size_t)m * N + n0]      = (__bf16)v0;
      outBf[(size_t)m * N + n0 + 16] = (__bf16)v1;
    } else {
      outF[(size_t)m * N + n0]      = v0;
      outF[(size_t)m * N + n0 + 16] = v1;
    }
  }
}

// ---------------- BiLSTM recurrence ----------------
// grid = 2 (direction), block = 1024 (32 waves). One WGP per direction.
// LDS (dynamic, 312320 B): U^T[512][128] bf16 | z[64][512] f32 | c[64][128] f32
//                          | h[64][136] bf16 (pitch 136)
#define LSTM_LDS_U   0
#define LSTM_LDS_Z   131072
#define LSTM_LDS_C   262144
#define LSTM_LDS_H   294912
#define LSTM_LDS_TOT 312320
#define HPITCH 136

__global__ void lstm_kernel(const __bf16* __restrict__ Pf,
                            const __bf16* __restrict__ Pb,
                            const __bf16* __restrict__ UfT,
                            const __bf16* __restrict__ UbT,
                            const float* __restrict__ bf_,
                            const float* __restrict__ bb_,
                            const unsigned char* __restrict__ mask,
                            __bf16* __restrict__ hcat) {
  const int dir = blockIdx.x;
  const __bf16* P    = dir ? Pb  : Pf;
  const __bf16* Ut   = dir ? UbT : UfT;
  const float*  bias = dir ? bb_ : bf_;

  extern __shared__ char smem[];
  __bf16* UlsT = (__bf16*)(smem + LSTM_LDS_U);   // [G4H][H_] transposed
  float*  zls  = (float*)(smem + LSTM_LDS_Z);
  float*  cls  = (float*)(smem + LSTM_LDS_C);
  __bf16* hls  = (__bf16*)(smem + LSTM_LDS_H);

  // stage U^T (128KB bf16) into LDS once; zero state
  {
    const uint32_t* Usrc = (const uint32_t*)Ut;
    uint32_t* Udst = (uint32_t*)UlsT;
    for (int i = threadIdx.x; i < (H_ * G4H) / 2; i += 1024) Udst[i] = Usrc[i];
    for (int i = threadIdx.x; i < B_ * HPITCH; i += 1024) hls[i] = (__bf16)0.0f;
    for (int i = threadIdx.x; i < B_ * H_;     i += 1024) cls[i] = 0.f;
  }
  __syncthreads();

  const int w    = threadIdx.x >> 5;
  const int lane = threadIdx.x & 31;

  for (int step = 0; step < T_; ++step) {
    const int t = dir ? (T_ - 1 - step) : step;

    // z = h @ U : M=64, K=128, N=512 -> 128 tiles, 4 per wave, 4 k-steps each
#pragma unroll
    for (int q = 0; q < 4; ++q) {
      const int tIdx = w + 32 * q;
      const int mt = tIdx & 3;
      const int nt = tIdx >> 2;
      v8f acc = {0.f, 0.f, 0.f, 0.f, 0.f, 0.f, 0.f, 0.f};
#pragma unroll
      for (int kt = 0; kt < 4; ++kt) {
        v16bf a = load_frag_k1(hls + (size_t)(mt * 16) * HPITCH + kt * 32, HPITCH);
        v16bf b = load_frag_k1(UlsT + (size_t)(nt * 16) * H_ + kt * 32, H_);
        acc = __builtin_amdgcn_wmma_f32_16x16x32_bf16(false, a, false, b,
                                                      (short)0, acc, false, false);
      }
      const int n    = nt * 16 + (lane & 15);
      const int half = lane >> 4;
#pragma unroll
      for (int r = 0; r < 8; ++r)
        zls[(size_t)(mt * 16 + r + 8 * half) * G4H + n] = acc[r];
    }
    __syncthreads();

    // gate phase: 8192 (b,j) cells over 1024 threads
#pragma unroll
    for (int it = 0; it < 8; ++it) {
      const int p  = it * 1024 + threadIdx.x;
      const int bi = p >> 7;
      const int j  = p & 127;
      const size_t pbase = ((size_t)(bi * T_ + t)) * G4H;
      const float zi = zls[(size_t)bi * G4H + j]          + (float)P[pbase + j]          + bias[j];
      const float zf = zls[(size_t)bi * G4H + H_ + j]     + (float)P[pbase + H_ + j]     + bias[H_ + j];
      const float zg = zls[(size_t)bi * G4H + 2 * H_ + j] + (float)P[pbase + 2 * H_ + j] + bias[2 * H_ + j];
      const float zo = zls[(size_t)bi * G4H + 3 * H_ + j] + (float)P[pbase + 3 * H_ + j] + bias[3 * H_ + j];
      if (mask[bi * T_ + t]) {
        const float ig = 1.f / (1.f + __expf(-zi));
        const float fg = 1.f / (1.f + __expf(-zf));
        const float og = 1.f / (1.f + __expf(-zo));
        const float g  = tanhf(zg);
        float c = fg * cls[(size_t)bi * H_ + j] + ig * g;
        cls[(size_t)bi * H_ + j] = c;
        hls[(size_t)bi * HPITCH + j] = (__bf16)(og * tanhf(c));
      }
      hcat[((size_t)(bi * T_ + t)) * D_ + dir * H_ + j] = hls[(size_t)bi * HPITCH + j];
      if (it == 0) {  // prefetch next step's x@W slab (global_prefetch_b8)
        const int tn = dir ? (t - 1) : (t + 1);
        if ((unsigned)tn < (unsigned)T_)
          __builtin_prefetch(&P[((size_t)(bi * T_ + tn)) * G4H + j], 0, 1);
      }
    }
    __syncthreads();
  }
}

// ---------------- attention ----------------
// grid = (T/16, NH, B); block = 256 (8 waves). Per-block: 16 query rows, one head.
__global__ void attn_kernel(const __bf16* __restrict__ qb,
                            const __bf16* __restrict__ kb,
                            const __bf16* __restrict__ vT,  // [B,NH,KD,T]
                            const unsigned char* __restrict__ mask,
                            __bf16* __restrict__ ctxb) {
  const int qt = blockIdx.x;
  const int h  = blockIdx.y;
  const int b  = blockIdx.z;

  __shared__ __align__(16) float  S[16][T_];     // 32 KB scores/exp
  __shared__ __align__(16) __bf16 Abf[16][T_];   // 16 KB normalized attn (WMMA A)
  __shared__ float  ctxAcc[16][KD_];             // 2 KB
  __shared__ float  red[16][16];
  __shared__ float  rowred[16];

  const int tid  = threadIdx.x;
  const int w    = tid >> 5;
  const int lane = tid & 31;
  const float scale = 0.17677669529663687f;  // 1/sqrt(KD)

  ((float*)ctxAcc)[tid]       = 0.f;
  ((float*)ctxAcc)[tid + 256] = 0.f;

  // Q fragment: rows = 16 queries, K = KD (row stride D_)
  const __bf16* qbase = qb + ((size_t)(b * T_ + qt * 16)) * D_ + h * KD_;
  v16bf qa = load_frag_k1(qbase, D_);

  // scores = Q @ K^T * scale, masked -> S
#pragma unroll
  for (int jj = 0; jj < 4; ++jj) {
    const int jt = w + jj * 8;
    const __bf16* kbase = kb + ((size_t)(b * T_ + jt * 16)) * D_ + h * KD_;
    v16bf kf = load_frag_k1(kbase, D_);  // K^T fragment: lane->key pos, e->kd
    v8f s = {0.f, 0.f, 0.f, 0.f, 0.f, 0.f, 0.f, 0.f};
    s = __builtin_amdgcn_wmma_f32_16x16x32_bf16(false, qa, false, kf,
                                                (short)0, s, false, false);
    const int scol = jt * 16 + (lane & 15);
    const bool mv  = mask[b * T_ + scol] != 0;
    const int half = lane >> 4;
#pragma unroll
    for (int r = 0; r < 8; ++r)
      S[r + 8 * half][scol] = mv ? s[r] * scale : NEGF;
  }
  __syncthreads();

  // masked softmax over 512 keys per row (16 rows, 16 threads/row)
  const int rr = tid >> 4;
  const int ci = tid & 15;
  float pm = -3.0e38f;
  for (int k2 = 0; k2 < 32; ++k2) pm = fmaxf(pm, S[rr][ci + 16 * k2]);
  red[rr][ci] = pm;
  __syncthreads();
  if (ci == 0) {
    float m2 = -3.0e38f;
    for (int i = 0; i < 16; ++i) m2 = fmaxf(m2, red[rr][i]);
    rowred[rr] = m2;
  }
  __syncthreads();
  const float rmax = rowred[rr];
  float ps = 0.f;
  for (int k2 = 0; k2 < 32; ++k2) {
    const int c = ci + 16 * k2;
    const float e = __expf(S[rr][c] - rmax);
    S[rr][c] = e;
    ps += e;
  }
  red[rr][ci] = ps;
  __syncthreads();
  if (ci == 0) {
    float s2 = 0.f;
    for (int i = 0; i < 16; ++i) s2 += red[rr][i];
    rowred[rr] = s2;
  }
  __syncthreads();
  const float inv = 1.0f / rowred[rr];
  for (int k2 = 0; k2 < 32; ++k2) {
    const int c = ci + 16 * k2;
    Abf[rr][c] = (__bf16)(S[rr][c] * inv);
  }
  __syncthreads();

  // ctx = attn[16x512] @ V[512x32]; each wave does 2 of 16 k-chunks, LDS-reduce
  v8f c0 = {0.f, 0.f, 0.f, 0.f, 0.f, 0.f, 0.f, 0.f};
  v8f c1 = {0.f, 0.f, 0.f, 0.f, 0.f, 0.f, 0.f, 0.f};
  const __bf16* vbase = vT + (size_t)(b * NH_ + h) * KD_ * T_;
#pragma unroll
  for (int kk2 = 0; kk2 < 2; ++kk2) {
    const int kt = w + kk2 * 8;
    v16bf aa = load_frag_k1(&Abf[0][kt * 32], T_);
    v16bf b0 = load_frag_k1(vbase + kt * 32, T_);                     // kd 0..15
    v16bf b1 = load_frag_k1(vbase + (size_t)16 * T_ + kt * 32, T_);   // kd 16..31
    c0 = __builtin_amdgcn_wmma_f32_16x16x32_bf16(false, aa, false, b0,
                                                 (short)0, c0, false, false);
    c1 = __builtin_amdgcn_wmma_f32_16x16x32_bf16(false, aa, false, b1,
                                                 (short)0, c1, false, false);
  }
  const int half2 = lane >> 4;
  const int ncol  = lane & 15;
#pragma unroll
  for (int r = 0; r < 8; ++r) {
    atomicAdd(&ctxAcc[r + 8 * half2][ncol],      c0[r]);   // ds_add_f32
    atomicAdd(&ctxAcc[r + 8 * half2][ncol + 16], c1[r]);
  }
  __syncthreads();

  for (int i = tid; i < 512; i += 256) {
    const int m = i >> 5, n = i & 31;
    ctxb[((size_t)(b * T_ + qt * 16 + m)) * D_ + h * KD_ + n] = (__bf16)ctxAcc[m][n];
  }
}

// ---------------- masked avg/max pooling ----------------
__global__ void pool_kernel(const float* __restrict__ att,
                            const unsigned char* __restrict__ mask,
                            __bf16* __restrict__ pbf) {
  const int b = blockIdx.x;
  const int d = threadIdx.x;  // 0..255
  float s = 0.f, mx = NEGF;
  int cnt = 0;
  for (int t = 0; t < T_; ++t) {
    const float v = att[((size_t)(b * T_ + t)) * D_ + d];
    if (mask[b * T_ + t]) { s += v; ++cnt; mx = fmaxf(mx, v); }
  }
  const float avg = s / fmaxf((float)cnt, 1.0f);
  pbf[(size_t)b * (2 * D_) + d]      = (__bf16)avg;
  pbf[(size_t)b * (2 * D_) + D_ + d] = (__bf16)mx;
}

// ---------------- host orchestration ----------------
extern "C" void kernel_launch(void* const* d_in, const int* in_sizes, int n_in,
                              void* d_out, int out_size, void* d_ws, size_t ws_size,
                              hipStream_t stream) {
  (void)in_sizes; (void)n_in; (void)out_size; (void)ws_size;
  const int*   seq = (const int*)  d_in[0];
  const float* emb = (const float*)d_in[1];
  const float* Wf  = (const float*)d_in[2];
  const float* Uf  = (const float*)d_in[3];
  const float* bf  = (const float*)d_in[4];
  const float* Wb  = (const float*)d_in[5];
  const float* Ub  = (const float*)d_in[6];
  const float* bb  = (const float*)d_in[7];
  const float* Wq  = (const float*)d_in[8];
  const float* bq  = (const float*)d_in[9];
  const float* Wk  = (const float*)d_in[10];
  const float* bk  = (const float*)d_in[11];
  const float* Wv  = (const float*)d_in[12];
  const float* bv  = (const float*)d_in[13];
  const float* Wo  = (const float*)d_in[14];
  const float* bo  = (const float*)d_in[15];
  const float* W1  = (const float*)d_in[16];
  const float* b1  = (const float*)d_in[17];
  const float* W2  = (const float*)d_in[18];
  const float* b2  = (const float*)d_in[19];
  const float* W3  = (const float*)d_in[20];
  const float* b3  = (const float*)d_in[21];

  size_t off = 0;
  auto alloc = [&](size_t bytes) -> void* {
    void* p = (char*)d_ws + off;
    off += (bytes + 255) & ~(size_t)255;
    return p;
  };
  unsigned char* maskbuf = (unsigned char*)alloc((size_t)B_ * T_);
  __bf16* xbf  = (__bf16*)alloc((size_t)B_ * T_ * E_ * 2);
  __bf16* WfT  = (__bf16*)alloc((size_t)E_ * G4H * 2);
  __bf16* WbT  = (__bf16*)alloc((size_t)E_ * G4H * 2);
  __bf16* UfT  = (__bf16*)alloc((size_t)H_ * G4H * 2);
  __bf16* UbT  = (__bf16*)alloc((size_t)H_ * G4H * 2);
  __bf16* WqT  = (__bf16*)alloc((size_t)D_ * D_ * 2);
  __bf16* WkT  = (__bf16*)alloc((size_t)D_ * D_ * 2);
  __bf16* WvT  = (__bf16*)alloc((size_t)D_ * D_ * 2);
  __bf16* WoT  = (__bf16*)alloc((size_t)D_ * D_ * 2);
  __bf16* W1T  = (__bf16*)alloc((size_t)(2 * D_) * DU_ * 2);
  __bf16* W2T  = (__bf16*)alloc((size_t)DU_ * (DU_ / 2) * 2);
  __bf16* W3T  = (__bf16*)alloc((size_t)(DU_ / 2) * OUT_ * 2);
  __bf16* Pf   = (__bf16*)alloc((size_t)B_ * T_ * G4H * 2);
  __bf16* Pb   = (__bf16*)alloc((size_t)B_ * T_ * G4H * 2);
  __bf16* hcat = (__bf16*)alloc((size_t)B_ * T_ * D_ * 2);
  __bf16* qbuf = (__bf16*)alloc((size_t)B_ * T_ * D_ * 2);
  __bf16* kbuf = (__bf16*)alloc((size_t)B_ * T_ * D_ * 2);
  __bf16* vbuf = (__bf16*)alloc((size_t)B_ * T_ * D_ * 2);
  __bf16* vT   = (__bf16*)alloc((size_t)B_ * T_ * D_ * 2);
  __bf16* ctxb = (__bf16*)alloc((size_t)B_ * T_ * D_ * 2);
  float*  attf = (float*) alloc((size_t)B_ * T_ * D_ * 4);
  __bf16* pbf  = (__bf16*)alloc((size_t)B_ * 2 * D_ * 2);
  __bf16* x1bf = (__bf16*)alloc((size_t)B_ * DU_ * 2);
  __bf16* x2bf = (__bf16*)alloc((size_t)B_ * (DU_ / 2) * 2);

  auto cvtT = [&](const float* s, __bf16* d, int K, int N) {
    cvt_transpose_kernel<<<(K * N + 255) / 256, 256, 0, stream>>>(s, d, K, N);
  };
  cvtT(Wf, WfT, E_, G4H);   cvtT(Wb, WbT, E_, G4H);
  cvtT(Uf, UfT, H_, G4H);   cvtT(Ub, UbT, H_, G4H);
  cvtT(Wq, WqT, D_, D_);    cvtT(Wk, WkT, D_, D_);
  cvtT(Wv, WvT, D_, D_);    cvtT(Wo, WoT, D_, D_);
  cvtT(W1, W1T, 2 * D_, DU_);
  cvtT(W2, W2T, DU_, DU_ / 2);
  cvtT(W3, W3T, DU_ / 2, OUT_);

  embed_kernel<<<B_ * T_, E_, 0, stream>>>(seq, emb, xbf, maskbuf);

  const int M = B_ * T_;  // 32768
  // precompute x@W for both directions (bias folded into gate phase instead)
  gemm_bf16_kernel<<<dim3(G4H / 256, M / 16), 256, 0, stream>>>(
      xbf, WfT, nullptr, Pf, nullptr, M, G4H, E_, 0);
  gemm_bf16_kernel<<<dim3(G4H / 256, M / 16), 256, 0, stream>>>(
      xbf, WbT, nullptr, Pb, nullptr, M, G4H, E_, 0);

  // recurrence: U^T + state resident in 312KB LDS, one WGP per direction
  hipFuncSetAttribute((const void*)lstm_kernel,
                      hipFuncAttributeMaxDynamicSharedMemorySize, LSTM_LDS_TOT);
  lstm_kernel<<<2, 1024, LSTM_LDS_TOT, stream>>>(Pf, Pb, UfT, UbT, bf, bb,
                                                 maskbuf, hcat);

  // QKV projections
  gemm_bf16_kernel<<<dim3(1, M / 16), 256, 0, stream>>>(
      hcat, WqT, bq, qbuf, nullptr, M, D_, D_, 0);
  gemm_bf16_kernel<<<dim3(1, M / 16), 256, 0, stream>>>(
      hcat, WkT, bk, kbuf, nullptr, M, D_, D_, 0);
  gemm_bf16_kernel<<<dim3(1, M / 16), 256, 0, stream>>>(
      hcat, WvT, bv, vbuf, nullptr, M, D_, D_, 0);

  vtrans_kernel<<<(B_ * T_ * D_) / 256, 256, 0, stream>>>(vbuf, vT);

  attn_kernel<<<dim3(T_ / 16, NH_, B_), 256, 0, stream>>>(qbuf, kbuf, vT,
                                                          maskbuf, ctxb);

  // output projection (fp32 out for pooling)
  gemm_bf16_kernel<<<dim3(1, M / 16), 256, 0, stream>>>(
      ctxb, WoT, bo, nullptr, attf, M, D_, D_, 0);

  pool_kernel<<<B_, D_, 0, stream>>>(attf, maskbuf, pbf);

  // MLP head
  gemm_bf16_kernel<<<dim3(1, B_ / 16), 256, 0, stream>>>(
      pbf, W1T, b1, x1bf, nullptr, B_, DU_, 2 * D_, 1);
  gemm_bf16_kernel<<<dim3(1, B_ / 16), 256, 0, stream>>>(
      x1bf, W2T, b2, x2bf, nullptr, B_, DU_ / 2, DU_, 1);
  gemm_bf16_kernel<<<dim3(1, B_ / 16), 256, 0, stream>>>(
      x2bf, W3T, b3, nullptr, (float*)d_out, B_, OUT_, DU_ / 2, 0);
}
